// PARALLELNEURONLAYER_57492432224774
// MI455X (gfx1250) — matrix operand order repaired and checked by reading, not compiled, not run on hardware
//
#include <hip/hip_runtime.h>
#include <cmath>

// ---------------------------------------------------------------------------
// out[s,n] = tanh( sum_d x[s,d] * W[n,d] + b[n] )
//   x: [8192,1024] f32 row-major, W: [4096,1024] f32 row-major, b: [4096]
// bf16x3 split GEMM on v_wmma_f32_16x16x32_bf16 (fp32-class accuracy).
// Block tile 256x128 (8 wave32s), wave tile 64x64 (16 accumulators),
// 48 WMMA : 32 ds_load_b128 per wave per K-step.
// ---------------------------------------------------------------------------

typedef unsigned short ushort_t;
typedef __attribute__((ext_vector_type(16))) __bf16 v16bf;
typedef __attribute__((ext_vector_type(8)))  float  v8f;

#define SEQ  8192
#define DIMK 1024
#define NOUT 4096

#define BM 256
#define BN 128
#define BK 32
#define LSTR 40          // LDS row stride in ushorts: 80B rows, 16B aligned, padded

struct QPair { uint4 a, b; };
union FragCast { QPair q; v16bf v; };

// Split fp32 -> bf16 hi (truncated) + bf16 lo (residual), pack 4 at a time.
__device__ __forceinline__ void storeSplit4(const float4 f, ushort_t* hp, ushort_t* lp) {
    unsigned u[4] = {__float_as_uint(f.x), __float_as_uint(f.y),
                     __float_as_uint(f.z), __float_as_uint(f.w)};
    unsigned h[4], l[4];
#pragma unroll
    for (int j = 0; j < 4; ++j) {
        h[j] = u[j] >> 16;
        float fv = __uint_as_float(u[j]);
        float fh = __uint_as_float(u[j] & 0xffff0000u);   // exact hi part
        l[j] = __float_as_uint(fv - fh) >> 16;            // residual, truncated to bf16
    }
    uint2 hv = make_uint2(h[0] | (h[1] << 16), h[2] | (h[3] << 16));
    uint2 lv = make_uint2(l[0] | (l[1] << 16), l[2] | (l[3] << 16));
    *(uint2*)hp = hv;
    *(uint2*)lp = lv;
}

// A operand (16x32, M x K): lane<16 -> M=lane, K0-7 & K16-23 ; lane>=16 -> K8-15 & K24-31
__device__ __forceinline__ v16bf loadFragA(const ushort_t* rowp, int halfSel) {
    FragCast f;
    f.q.a = *(const uint4*)(rowp + halfSel * 8);
    f.q.b = *(const uint4*)(rowp + halfSel * 8 + 16);
    return f.v;
}

// B operand (32x16, K x N): lane<16 -> N=lane, K0-7 & K16-23 ; lane>=16 -> K8-15 & K24-31.
// B[k][n] == W[n][k], so this is a contiguous read of W's row n.
__device__ __forceinline__ v16bf loadFragB(const ushort_t* rowp, int halfSel) {
    FragCast f;
    f.q.a = *(const uint4*)(rowp + halfSel * 8);
    f.q.b = *(const uint4*)(rowp + halfSel * 8 + 16);
    return f.v;
}

// Branch-free tanh: native v_tanh_f32 when available, else exp2/rcp trans ops.
__device__ __forceinline__ float fast_tanh(float x) {
#if __has_builtin(__builtin_amdgcn_tanhf)
    return __builtin_amdgcn_tanhf(x);
#else
    // tanh(x) = 1 - 2/(exp(2x)+1); exp2/rcp saturate correctly at +/-inf.
    const float t = __builtin_amdgcn_exp2f(x * 2.8853900817779268f); // 2*log2(e)
    return 1.0f - 2.0f * __builtin_amdgcn_rcpf(t + 1.0f);
#endif
}

__global__ __launch_bounds__(256)
void PARALLELNEURONLAYER_57492432224774_kernel(const float* __restrict__ X,
                                               const float* __restrict__ W,
                                               const float* __restrict__ Bv,
                                               float* __restrict__ OUT) {
    __shared__ __align__(16) ushort_t s_xh[BM * LSTR];   // 20 KB
    __shared__ __align__(16) ushort_t s_xl[BM * LSTR];   // 20 KB
    __shared__ __align__(16) ushort_t s_wh[BN * LSTR];   // 10 KB
    __shared__ __align__(16) ushort_t s_wl[BN * LSTR];   // 10 KB

    const int tid     = threadIdx.x;
    const int lane    = tid & 31;
    const int wid     = tid >> 5;       // 8 wave32s
    const int halfSel = lane >> 4;      // 0: lanes 0-15, 1: lanes 16-31
    const int lrow    = lane & 15;

    const int waveM = wid & 3;          // 4 waves along M: 64 rows each
    const int waveN = wid >> 2;         // 2 waves along N: 64 cols each

    const int mBlock = blockIdx.y * BM;
    const int nBlock = blockIdx.x * BN;

    // staging geometry: 8 consecutive threads cover one 128B K-row.
    const int ldRow = tid >> 3;         // 0..31
    const int ldCol = (tid & 7) * 4;    // float offset within K-tile

    v8f acc[4][4];
#pragma unroll
    for (int mt = 0; mt < 4; ++mt)
#pragma unroll
        for (int nt = 0; nt < 4; ++nt)
            acc[mt][nt] = (v8f)(0.0f);

    for (int kb = 0; kb < DIMK / BK; ++kb) {
        const int k0 = kb * BK;

        // Global fp32 -> registers (issues early, overlaps prior compute)
        float4 xr[8], wr[4];
#pragma unroll
        for (int i = 0; i < 8; ++i) {
            const int row = ldRow + i * 32;           // 0..255
            xr[i] = *(const float4*)(X + (size_t)(mBlock + row) * DIMK + k0 + ldCol);
        }
#pragma unroll
        for (int i = 0; i < 4; ++i) {
            const int row = ldRow + i * 32;           // 0..127
            wr[i] = *(const float4*)(W + (size_t)(nBlock + row) * DIMK + k0 + ldCol);
        }

        __syncthreads();    // previous iteration's LDS reads complete
#pragma unroll
        for (int i = 0; i < 8; ++i) {
            const int row = ldRow + i * 32;
            storeSplit4(xr[i], &s_xh[row * LSTR + ldCol], &s_xl[row * LSTR + ldCol]);
        }
#pragma unroll
        for (int i = 0; i < 4; ++i) {
            const int row = ldRow + i * 32;
            storeSplit4(wr[i], &s_wh[row * LSTR + ldCol], &s_wl[row * LSTR + ldCol]);
        }
        __syncthreads();    // tile visible to all waves

        // A fragments for this wave's 4 M-tiles (hi+lo)
        v16bf ah[4], al[4];
#pragma unroll
        for (int mt = 0; mt < 4; ++mt) {
            const int r = waveM * 64 + mt * 16 + lrow;
            ah[mt] = loadFragA(&s_xh[r * LSTR], halfSel);
            al[mt] = loadFragA(&s_xl[r * LSTR], halfSel);
        }

        // Stream B fragments per N-tile; bf16x3: Ahi*Bhi + Ahi*Blo + Alo*Bhi
#pragma unroll
        for (int nt = 0; nt < 4; ++nt) {
            const int r = waveN * 64 + nt * 16 + lrow;
            const v16bf bh = loadFragB(&s_wh[r * LSTR], halfSel);
            const v16bf bl = loadFragB(&s_wl[r * LSTR], halfSel);
#pragma unroll
            for (int mt = 0; mt < 4; ++mt) {
                acc[mt][nt] = __builtin_amdgcn_wmma_f32_16x16x32_bf16(
                    false, ah[mt], false, bh, (short)0, acc[mt][nt], false, false);
                acc[mt][nt] = __builtin_amdgcn_wmma_f32_16x16x32_bf16(
                    false, ah[mt], false, bl, (short)0, acc[mt][nt], false, false);
                acc[mt][nt] = __builtin_amdgcn_wmma_f32_16x16x32_bf16(
                    false, al[mt], false, bh, (short)0, acc[mt][nt], false, false);
            }
        }
    }

    // Epilogue: C/D layout -> lane<16: N=lane, M=VGPRr ; lane>=16: N=lane-16, M=8+r
#pragma unroll
    for (int nt = 0; nt < 4; ++nt) {
        const int col  = nBlock + waveN * 64 + nt * 16 + lrow;
        const float bias = Bv[col];
#pragma unroll
        for (int mt = 0; mt < 4; ++mt) {
            const int rowBase = mBlock + waveM * 64 + mt * 16 + halfSel * 8;
#pragma unroll
            for (int r = 0; r < 8; ++r) {
                const float v = acc[mt][nt][r] + bias;
                OUT[(size_t)(rowBase + r) * NOUT + col] = fast_tanh(v);
            }
        }
    }
}

extern "C" void kernel_launch(void* const* d_in, const int* in_sizes, int n_in,
                              void* d_out, int out_size, void* d_ws, size_t ws_size,
                              hipStream_t stream) {
    (void)in_sizes; (void)n_in; (void)out_size; (void)d_ws; (void)ws_size;
    const float* x = (const float*)d_in[0];
    const float* W = (const float*)d_in[1];
    const float* b = (const float*)d_in[2];
    float* out = (float*)d_out;

    dim3 grid(NOUT / BN, SEQ / BM);   // 32 x 32 = 1024 workgroups
    dim3 block(256);                  // 8 wave32s
    PARALLELNEURONLAYER_57492432224774_kernel<<<grid, block, 0, stream>>>(x, W, b, out);
}
